// LoRALinear1d_52158082843371
// MI455X (gfx1250) — matrix operand order repaired and checked by reading, not compiled, not run on hardware
//
#include <hip/hip_runtime.h>
#include <hip/hip_bf16.h>
#include <stdint.h>

// Problem constants (from reference)
#define B_N   8
#define CIN_  256
#define COUT_ 256
#define L_    32768
// Tiling
#define NT    128          // l-columns per block
#define SROW  264          // LDS row stride in bf16 elements (16B-aligned rows, +16B pad)

typedef __attribute__((ext_vector_type(16))) __bf16 v16bf;
typedef __attribute__((ext_vector_type(8)))  float  v8f;

union ABf {            // 32 bytes: two 16B chunks viewed as a WMMA 16-element bf16 fragment
  uint4 u[2];
  v16bf v;
};

__device__ __forceinline__ unsigned short f2bf(float f) {
  unsigned u = __float_as_uint(f);
  u += 0x7FFFu + ((u >> 16) & 1u);          // round-to-nearest-even
  return (unsigned short)(u >> 16);
}
__device__ __forceinline__ float bf2f(unsigned short h) {
  return __uint_as_float(((unsigned)h) << 16);
}

// ---------------------------------------------------------------------------
// Kernel 1: fold LoRA rank-2 update into per-batch effective weights,
// emit bf16 hi/lo split pair. One block per batch.
// ---------------------------------------------------------------------------
__global__ __launch_bounds__(256)
void prep_weff(const float* __restrict__ g_out,   // (B, CINFO, 1)
               const float* __restrict__ W_main,  // (COUT, CIN)
               const float* __restrict__ W_ain,   // (CIN*R, CINFO), k = i*2 + r
               const float* __restrict__ W_aout,  // (COUT*R, CINFO), k = r*COUT + o
               unsigned short* __restrict__ Whi,  // (B, COUT, CIN) bf16
               unsigned short* __restrict__ Wlo)  // (B, COUT, CIN) bf16
{
  const int b = blockIdx.x;
  const int t = threadIdx.x;

  __shared__ float gs[CIN_];
  __shared__ float ain0[CIN_], ain1[CIN_];
  __shared__ float aout0[COUT_], aout1[COUT_];

  gs[t] = g_out[b * CIN_ + t];
  __syncthreads();

  // a_in[b, t, r] and a_out[b, r, t]
  float s0 = 0.f, s1 = 0.f, s2 = 0.f, s3 = 0.f;
  const float* wa0 = W_ain  + (size_t)(t * 2 + 0) * CIN_;
  const float* wa1 = W_ain  + (size_t)(t * 2 + 1) * CIN_;
  const float* wb0 = W_aout + (size_t)(0 * COUT_ + t) * CIN_;
  const float* wb1 = W_aout + (size_t)(1 * COUT_ + t) * CIN_;
  #pragma unroll 4
  for (int c = 0; c < CIN_; ++c) {
    const float gv = gs[c];
    s0 += wa0[c] * gv;  s1 += wa1[c] * gv;
    s2 += wb0[c] * gv;  s3 += wb1[c] * gv;
  }
  ain0[t] = s0;  ain1[t] = s1;  aout0[t] = s2;  aout1[t] = s3;
  __syncthreads();

  // W_eff[b, o=t, i] = W_main[o,i] + aout0[o]*ain0[i] + aout1[o]*ain1[i]
  const float a0 = aout0[t], a1 = aout1[t];
  const float* wm = W_main + (size_t)t * CIN_;
  unsigned short* wh = Whi + ((size_t)b * COUT_ + t) * CIN_;
  unsigned short* wl = Wlo + ((size_t)b * COUT_ + t) * CIN_;
  #pragma unroll 4
  for (int i = 0; i < CIN_; ++i) {
    const float w = wm[i] + a0 * ain0[i] + a1 * ain1[i];
    const unsigned short h = f2bf(w);
    wh[i] = h;
    wl[i] = f2bf(w - bf2f(h));
  }
}

// ---------------------------------------------------------------------------
// Kernel 2: out[b,o,l] = W_eff[b]·x[b] + bias, bf16 hi/lo split WMMA.
// Grid: (L/NT, B). Block: 256 threads = 8 waves; wave w owns M rows
// [w*32, w*32+32) (2 M-tiles) x all 8 N-subtiles of the 128-wide l tile.
// ---------------------------------------------------------------------------
__global__ __launch_bounds__(256)
void gemm_lora(const float* __restrict__ x,       // (B, CIN, L)
               const float* __restrict__ b_main,  // (COUT)
               const unsigned short* __restrict__ Whi,
               const unsigned short* __restrict__ Wlo,
               float* __restrict__ out)           // (B, COUT, L)
{
  extern __shared__ unsigned short smem[];        // 2 * NT * SROW halves = 132KB
  unsigned short* __restrict__ xh = smem;                 // transposed [l][c] hi
  unsigned short* __restrict__ xl = smem + NT * SROW;     // transposed [l][c] lo

  const int b  = blockIdx.y;
  const int l0 = blockIdx.x * NT;
  const int t  = threadIdx.x;
  const float* xb = x + (size_t)b * CIN_ * L_;

  // ---- Stage x tile: 256 (c) x 128 (l) fp32 -> LDS transposed bf16 hi/lo ----
  #pragma unroll 4
  for (int j = 0; j < 32; ++j) {
    const int id = j * 256 + t;
    const int c  = id >> 5;          // 0..255
    const int l  = (id & 31) * 4;    // 0..124
    const float4 v4 = *(const float4*)(xb + (size_t)c * L_ + l0 + l);
    float f[4] = {v4.x, v4.y, v4.z, v4.w};
    #pragma unroll
    for (int i = 0; i < 4; ++i) {
      const unsigned short h = f2bf(f[i]);
      xh[(l + i) * SROW + c] = h;
      xl[(l + i) * SROW + c] = f2bf(f[i] - bf2f(h));
    }
  }
  __syncthreads();

  const int wave  = t >> 5;         // 0..7
  const int lane  = t & 31;
  const int lrow  = lane & 15;      // row within 16-lane half
  const int khalf = lane >> 4;      // 0/1 -> which K-half this lane holds
  const int m0    = wave * 32;      // 32 output rows per wave

  v8f acc[8][2];
  #pragma unroll
  for (int n = 0; n < 8; ++n)
    #pragma unroll
    for (int m = 0; m < 2; ++m)
      #pragma unroll
      for (int j = 0; j < 8; ++j)
        acc[n][m][j] = 0.f;

  // ---- K loop: 256 in steps of 32 ----
  for (int k = 0; k < CIN_; k += 32) {
    // A fragments: per the 16-bit 16x32 A layout, lane (lrow,khalf) needs
    // K = k + khalf*8 + [0..7] in v0-3 and +16 in v4-7. Two aligned 16B loads.
    ABf ah[2], al[2];
    #pragma unroll
    for (int m = 0; m < 2; ++m) {
      const size_t row = (size_t)b * COUT_ + (m0 + m * 16 + lrow);
      const unsigned short* ph = Whi + row * CIN_ + k + khalf * 8;
      const unsigned short* pl = Wlo + row * CIN_ + k + khalf * 8;
      ah[m].u[0] = *(const uint4*)(ph);
      ah[m].u[1] = *(const uint4*)(ph + 16);
      al[m].u[0] = *(const uint4*)(pl);
      al[m].u[1] = *(const uint4*)(pl + 16);
    }
    #pragma unroll
    for (int n = 0; n < 8; ++n) {
      // B fragment: column N = lrow holds K = k + khalf*8 + [0..7] and +16,
      // matching A -> two aligned ds_load_b128 from the transposed tile.
      const int lloc = n * 16 + lrow;
      const unsigned short* bh = &xh[lloc * SROW + k + khalf * 8];
      const unsigned short* bl = &xl[lloc * SROW + k + khalf * 8];
      ABf bhF, blF;
      bhF.u[0] = *(const uint4*)(bh);
      bhF.u[1] = *(const uint4*)(bh + 16);
      blF.u[0] = *(const uint4*)(bl);
      blF.u[1] = *(const uint4*)(bl + 16);
      #pragma unroll
      for (int m = 0; m < 2; ++m) {
        // hi*hi + hi*lo + lo*hi  (lo*lo ~2^-32, dropped)
        acc[n][m] = __builtin_amdgcn_wmma_f32_16x16x32_bf16(
            false, ah[m].v, false, bhF.v, (short)0, acc[n][m], false, false);
        acc[n][m] = __builtin_amdgcn_wmma_f32_16x16x32_bf16(
            false, ah[m].v, false, blF.v, (short)0, acc[n][m], false, false);
        acc[n][m] = __builtin_amdgcn_wmma_f32_16x16x32_bf16(
            false, al[m].v, false, bhF.v, (short)0, acc[n][m], false, false);
      }
    }
  }

  // ---- Epilogue: add bias, store f32. C/D layout: VGPR j -> M = khalf*8 + j. ----
  #pragma unroll
  for (int m = 0; m < 2; ++m) {
    float bias[8];
    #pragma unroll
    for (int j = 0; j < 8; ++j)
      bias[j] = b_main[m0 + m * 16 + khalf * 8 + j];
    #pragma unroll
    for (int n = 0; n < 8; ++n) {
      #pragma unroll
      for (int j = 0; j < 8; ++j) {
        const int o = m0 + m * 16 + khalf * 8 + j;
        out[((size_t)b * COUT_ + o) * L_ + l0 + n * 16 + lrow] = acc[n][m][j] + bias[j];
      }
    }
  }
}

// ---------------------------------------------------------------------------
extern "C" void kernel_launch(void* const* d_in, const int* in_sizes, int n_in,
                              void* d_out, int out_size, void* d_ws, size_t ws_size,
                              hipStream_t stream) {
  const float* x      = (const float*)d_in[0];
  const float* g_out  = (const float*)d_in[1];
  const float* W_main = (const float*)d_in[2];
  const float* b_main = (const float*)d_in[3];
  const float* W_ain  = (const float*)d_in[4];
  const float* W_aout = (const float*)d_in[5];

  unsigned short* Whi = (unsigned short*)d_ws;                       // 1 MB
  unsigned short* Wlo = Whi + (size_t)B_N * COUT_ * CIN_;            // 1 MB

  prep_weff<<<B_N, 256, 0, stream>>>(g_out, W_main, W_ain, W_aout, Whi, Wlo);

  dim3 grid(L_ / NT, B_N);
  const size_t smem = (size_t)2 * NT * SROW * sizeof(unsigned short); // 135168 B
  gemm_lora<<<grid, 256, smem, stream>>>(x, b_main, Whi, Wlo, (float*)d_out);
}
